// architecture_26594437497431
// MI455X (gfx1250) — compile-verified, matrix operand order checked by reading, not run on hardware
//
#include <hip/hip_runtime.h>
#include <hip/hip_bf16.h>
#include <cstdint>
#include <cstddef>

// ---------------------------------------------------------------------------
// Problem constants (match reference)
// ---------------------------------------------------------------------------
static constexpr int BB  = 64;
static constexpr int N1c = 4328, N2c = 1024, N3c = 256, N4c = 32, N5c = 2;
static constexpr int E1c = 51936, E2c = 16384, E3c = 4096, E4c = 256;
static constexpr int KNN = 3;

static inline int SP(int c) { return c < 32 ? 32 : c; }  // padded channel stride

__device__ static const float d_SHIFT[4] = {0.5f, -2.0f, -1.0f, 1.0f};
__device__ static const float d_SCALE[4] = {0.5f, 1.5f, 0.75f, 3.5f};

typedef __attribute__((ext_vector_type(16))) _Float16 v16h;
typedef __attribute__((ext_vector_type(8)))  _Float16 v8h;
typedef __attribute__((ext_vector_type(8)))  float    v8f;
typedef __attribute__((ext_vector_type(4)))  float    v4f;

#define DEVINL __device__ __forceinline__

// ---------------------------------------------------------------------------
// Elementwise / scatter kernels (stride-aware: C = logical, Cs = storage)
// ---------------------------------------------------------------------------
__global__ void k_normalize_pad(const float* __restrict__ x, float* __restrict__ o, int total) {
  int i = blockIdx.x * blockDim.x + threadIdx.x;   // total = B*N1*32
  if (i >= total) return;
  int c = i & 31, n = i >> 5;
  float v = 0.0f;
  if (c < 4) v = (x[n * 4 + c] - d_SHIFT[c]) / d_SCALE[c];
  o[i] = v;
}

__global__ void k_denorm(const float* __restrict__ h, float* __restrict__ o, int total) {
  int i = blockIdx.x * blockDim.x + threadIdx.x;   // total = B*N1*4, h has stride 32
  if (i >= total) return;
  int c = i & 3, n = i >> 2;
  o[i] = h[n * 32 + c] * d_SCALE[c] + d_SHIFT[c];
}

__global__ void k_zero(float* __restrict__ p, int n) {
  int i = blockIdx.x * blockDim.x + threadIdx.x;
  if (i < n) p[i] = 0.0f;
}

__global__ void k_deg(const int* __restrict__ dst, int E, float* __restrict__ deg) {
  int i = blockIdx.x * blockDim.x + threadIdx.x;
  if (i < E) atomicAdd(&deg[dst[i]], 1.0f);
}

__global__ void k_edge_scatter(const float* __restrict__ h, float* __restrict__ agg,
                               const int* __restrict__ src, const int* __restrict__ dst,
                               int E, int N, int C, int Cs, int total) {
  int i = blockIdx.x * blockDim.x + threadIdx.x;   // total = E*B*C
  if (i >= total) return;
  int c = i % C; int t = i / C; int b = t % BB; int e = t / BB;
  float v = h[((size_t)b * N + src[e]) * Cs + c];
  atomicAdd(&agg[((size_t)b * N + dst[e]) * Cs + c], v);
}

__global__ void k_div_deg(float* __restrict__ agg, const float* __restrict__ deg,
                          int N, int C, int Cs, int total) {
  int i = blockIdx.x * blockDim.x + threadIdx.x;   // total = B*N*C
  if (i >= total) return;
  int c = i % C; int t = i / C; int n = t % N; int b = t / N;
  agg[((size_t)b * N + n) * Cs + c] /= fmaxf(deg[n], 1.0f);
}

__global__ void k_label_cnt(const int* __restrict__ labels, int Nin, float* __restrict__ cnt) {
  int i = blockIdx.x * blockDim.x + threadIdx.x;
  if (i < Nin) atomicAdd(&cnt[labels[i]], 1.0f);
}

__global__ void k_pool_scatter(const float* __restrict__ h, float* __restrict__ o,
                               const int* __restrict__ labels,
                               int Nin, int Nout, int C, int Cs, int total) {
  int i = blockIdx.x * blockDim.x + threadIdx.x;   // total = B*Nin*C
  if (i >= total) return;
  int c = i % C; int t = i / C; int n = t % Nin; int b = t / Nin;
  atomicAdd(&o[((size_t)b * Nout + labels[n]) * Cs + c],
            h[((size_t)b * Nin + n) * Cs + c]);
}

__global__ void k_div_cnt(float* __restrict__ o, const float* __restrict__ cnt,
                          int Nout, int C, int Cs, int total) {
  int i = blockIdx.x * blockDim.x + threadIdx.x;   // total = B*Nout*C
  if (i >= total) return;
  int c = i % C; int t = i / C; int n = t % Nout; int b = t / Nout;
  o[((size_t)b * Nout + n) * Cs + c] /= fmaxf(cnt[n], 1.0f);
}

__global__ void k_interp_w(const int* __restrict__ yidx, const int* __restrict__ xidx,
                           const float* __restrict__ posx, const float* __restrict__ posy,
                           int E, float* __restrict__ w, float* __restrict__ den) {
  int j = blockIdx.x * blockDim.x + threadIdx.x;
  if (j >= E) return;
  int yj = yidx[j], xj = xidx[j];
  float d0 = posx[xj * 3 + 0] - posy[yj * 3 + 0];
  float d1 = posx[xj * 3 + 1] - posy[yj * 3 + 1];
  float d2 = posx[xj * 3 + 2] - posy[yj * 3 + 2];
  float wv = 1.0f / fmaxf(d0 * d0 + d1 * d1 + d2 * d2, 1e-16f);
  w[j] = wv;
  atomicAdd(&den[yj], wv);
}

__global__ void k_interp_scatter(const float* __restrict__ h, float* __restrict__ num,
                                 const int* __restrict__ yidx, const int* __restrict__ xidx,
                                 const float* __restrict__ w,
                                 int Nx, int Ny, int C, int Cs, int total) {
  int i = blockIdx.x * blockDim.x + threadIdx.x;   // total = E*B*C
  if (i >= total) return;
  int c = i % C; int t = i / C; int b = t % BB; int j = t / BB;
  float v = h[((size_t)b * Nx + xidx[j]) * Cs + c] * w[j];
  atomicAdd(&num[((size_t)b * Ny + yidx[j]) * Cs + c], v);
}

__global__ void k_interp_div(float* __restrict__ num, const float* __restrict__ den,
                             int Ny, int C, int Cs, int total) {
  int i = blockIdx.x * blockDim.x + threadIdx.x;   // total = B*Ny*C
  if (i >= total) return;
  int c = i % C; int t = i / C; int n = t % Ny; int b = t / Ny;
  num[((size_t)b * Ny + n) * Cs + c] /= den[n];
}

// ---------------------------------------------------------------------------
// Weight packing: fp32 [Cout, Cin] -> zero-padded f16 [Np, Kp]
// ---------------------------------------------------------------------------
__global__ void k_pack_w(const float* __restrict__ W, _Float16* __restrict__ Wp,
                         int Cout, int Cin, int Kp, int total) {
  int i = blockIdx.x * blockDim.x + threadIdx.x;   // total = Np*Kp
  if (i >= total) return;
  int k = i % Kp; int o = i / Kp;
  float v = (o < Cout && k < Cin) ? W[(size_t)o * Cin + k] : 0.0f;
  Wp[i] = (_Float16)v;
}

// ---------------------------------------------------------------------------
// WMMA dual-GEMM:  out = act( A0 @ W0p^T + [A1 @ W1p^T] + [bias] )
//   A*: [M, Kp] fp32 (pad lanes zero), W*p: [Np, Kp] f16 zero-padded.
//   Kp, Np multiples of 32/16; M multiple of 16. No masking in the K-loop.
//   One wave32 per 16x16 output tile, 8 waves per block.
// ---------------------------------------------------------------------------
DEVINL v16h load_a_frag32(const float* __restrict__ arun, int kb) {
  // arun already offset by half*8; needs runs [kb, kb+8) and [kb+16, kb+24)
  v4f f0 = *(const v4f*)(arun + kb);
  v4f f1 = *(const v4f*)(arun + kb + 4);
  v4f f2 = *(const v4f*)(arun + kb + 16);
  v4f f3 = *(const v4f*)(arun + kb + 20);
  v16h a;
#pragma unroll
  for (int t = 0; t < 4; ++t) {
    a[t]      = (_Float16)f0[t];
    a[4 + t]  = (_Float16)f1[t];
    a[8 + t]  = (_Float16)f2[t];
    a[12 + t] = (_Float16)f3[t];
  }
  return a;
}

DEVINL v16h load_b_frag16(const _Float16* __restrict__ wrun, int kb) {
  v8h b0 = *(const v8h*)(wrun + kb);
  v8h b1 = *(const v8h*)(wrun + kb + 16);
  return __builtin_shufflevector(b0, b1, 0, 1, 2, 3, 4, 5, 6, 7,
                                 8, 9, 10, 11, 12, 13, 14, 15);
}

__global__ __launch_bounds__(256)
void k_wmma_gemm(const float* __restrict__ A0, const _Float16* __restrict__ W0,
                 const float* __restrict__ A1, const _Float16* __restrict__ W1,
                 const float* __restrict__ bias, float* __restrict__ out,
                 int Mrows, int Kp, int Np, int Ncols, int applyElu) {
  const int lane = threadIdx.x & 31;
  const int wave = threadIdx.x >> 5;
  const int mTile = blockIdx.x * 8 + wave;
  if (mTile * 16 >= Mrows) return;            // uniform per-wave exit
  const int nTile = blockIdx.y;
  const int row0 = mTile * 16, col0 = nTile * 16;
  const int m = lane & 15, half = lane >> 4;

  __builtin_prefetch(W0, 0, 1);               // global_prefetch_b8 on weights
  if (W1) __builtin_prefetch(W1, 0, 1);

  const float*    a0run = A0 + (size_t)(row0 + m) * Kp + half * 8;
  const _Float16* w0run = W0 + (size_t)(col0 + m) * Kp + half * 8;

  v8f acc = {};
  for (int kb = 0; kb < Kp; kb += 32) {
    v16h a = load_a_frag32(a0run, kb);
    v16h b = load_b_frag16(w0run, kb);
    acc = __builtin_amdgcn_wmma_f32_16x16x32_f16(false, a, false, b,
                                                 (short)0, acc, false, false);
  }
  if (A1) {
    const float*    a1run = A1 + (size_t)(row0 + m) * Kp + half * 8;
    const _Float16* w1run = W1 + (size_t)(col0 + m) * Kp + half * 8;
    for (int kb = 0; kb < Kp; kb += 32) {
      v16h a = load_a_frag32(a1run, kb);
      v16h b = load_b_frag16(w1run, kb);
      acc = __builtin_amdgcn_wmma_f32_16x16x32_f16(false, a, false, b,
                                                   (short)0, acc, false, false);
    }
  }

  float bv = 0.0f;
  if (bias && (col0 + m) < Ncols) bv = bias[col0 + m];
#pragma unroll
  for (int j = 0; j < 8; ++j) {
    int r = row0 + half * 8 + j;
    float v = acc[j] + bv;
    if (applyElu) v = (v > 0.0f) ? v : (expf(v) - 1.0f);
    out[(size_t)r * Np + col0 + m] = v;       // pad cols store exact zeros
  }
}

// ---------------------------------------------------------------------------
// Host-side helpers
// ---------------------------------------------------------------------------
static inline int nblk(long long n) { return (int)((n + 255) / 256); }

static void zero_buf(float* p, long long n, hipStream_t s) {
  k_zero<<<nblk(n), 256, 0, s>>>(p, (int)n);
}

static void run_gemm(const float* A0, const _Float16* W0, const float* A1,
                     const _Float16* W1, const float* bias, float* out,
                     int M, int Kp, int Np, int Ncols, bool elu, hipStream_t s) {
  dim3 g((M / 16 + 7) / 8, Np / 16);
  k_wmma_gemm<<<g, 256, 0, s>>>(A0, W0, A1, W1, bias, out, M, Kp, Np, Ncols,
                                elu ? 1 : 0);
}

static void run_sage(const float* hin, float* agg, float* hout, const int* edge,
                     int E, int N, int Cin, int Cout, const _Float16* Wlp,
                     const _Float16* Wrp, float* deg, bool elu, hipStream_t s) {
  int Csi = SP(Cin), Np = SP(Cout);
  long long aggTot = (long long)BB * N * Csi;
  zero_buf(agg, aggTot, s);
  zero_buf(deg, N, s);
  const int* src = edge;
  const int* dst = edge + E;
  k_deg<<<nblk(E), 256, 0, s>>>(dst, E, deg);
  long long sTot = (long long)E * BB * Cin;
  k_edge_scatter<<<nblk(sTot), 256, 0, s>>>(hin, agg, src, dst, E, N, Cin, Csi, (int)sTot);
  long long dTot = (long long)BB * N * Cin;
  k_div_deg<<<nblk(dTot), 256, 0, s>>>(agg, deg, N, Cin, Csi, (int)dTot);
  run_gemm(agg, Wlp, hin, Wrp, nullptr, hout, BB * N, Csi, Np, Cout, elu, s);
}

static void run_pool(const float* hin, float* hout, const int* labels, int Nin,
                     int Nout, int C, float* cnt, hipStream_t s) {
  int Cs = SP(C);
  long long oPad = (long long)BB * Nout * Cs;
  zero_buf(hout, oPad, s);
  zero_buf(cnt, Nout, s);
  k_label_cnt<<<nblk(Nin), 256, 0, s>>>(labels, Nin, cnt);
  long long iTot = (long long)BB * Nin * C;
  k_pool_scatter<<<nblk(iTot), 256, 0, s>>>(hin, hout, labels, Nin, Nout, C, Cs, (int)iTot);
  long long oTot = (long long)BB * Nout * C;
  k_div_cnt<<<nblk(oTot), 256, 0, s>>>(hout, cnt, Nout, C, Cs, (int)oTot);
}

static void run_knn(const float* hin, float* hout, const int* ai, const float* posx,
                    const float* posy, int Nx, int Ny, int C, float* den, float* wbuf,
                    hipStream_t s) {
  int Cs = SP(C);
  int E = Ny * KNN;
  const int* yidx = ai;
  const int* xidx = ai + E;
  long long oPad = (long long)BB * Ny * Cs;
  zero_buf(hout, oPad, s);
  zero_buf(den, Ny, s);
  k_interp_w<<<nblk(E), 256, 0, s>>>(yidx, xidx, posx, posy, E, wbuf, den);
  long long sTot = (long long)E * BB * C;
  k_interp_scatter<<<nblk(sTot), 256, 0, s>>>(hin, hout, yidx, xidx, wbuf, Nx, Ny, C, Cs, (int)sTot);
  long long oTot = (long long)BB * Ny * C;
  k_interp_div<<<nblk(oTot), 256, 0, s>>>(hout, den, Ny, C, Cs, (int)oTot);
}

// ---------------------------------------------------------------------------
// Top-level launch
// ---------------------------------------------------------------------------
extern "C" void kernel_launch(void* const* d_in, const int* in_sizes, int n_in,
                              void* d_out, int out_size, void* d_ws, size_t ws_size,
                              hipStream_t stream) {
  const float* x    = (const float*)d_in[0];
  const int*   e1   = (const int*)d_in[1];
  const int*   e2   = (const int*)d_in[2];
  const int*   e3   = (const int*)d_in[3];
  const int*   e4   = (const int*)d_in[4];
  const int*   lab1 = (const int*)d_in[5];
  const int*   lab2 = (const int*)d_in[6];
  const int*   lab3 = (const int*)d_in[7];
  const int*   lab4 = (const int*)d_in[8];
  const float* pos1 = (const float*)d_in[9];
  const float* pos2 = (const float*)d_in[10];
  const float* pos3 = (const float*)d_in[11];
  const float* pos4 = (const float*)d_in[12];
  const float* pos5 = (const float*)d_in[13];
  const int*   ai54 = (const int*)d_in[14];
  const int*   ai43 = (const int*)d_in[15];
  const int*   ai32 = (const int*)d_in[16];
  const int*   ai21 = (const int*)d_in[17];
  const float* Wc[32];
  for (int i = 0; i < 32; ++i) Wc[i] = (const float*)d_in[18 + i];
  const float* W5 = (const float*)d_in[50];
  const float* b5 = (const float*)d_in[51];
  const float* W6 = (const float*)d_in[52];
  const float* b6 = (const float*)d_in[53];

  // Workspace layout
  const size_t CAP = (size_t)BB * N1c * 64;   // 17,727,488 floats per buffer
  float* bufA = (float*)d_ws;
  float* bufB = bufA + CAP;
  float* bufC = bufB + CAP;
  float* misc = bufC + CAP;
  float* deg  = misc;                  // up to N1 (4328); shared cnt/den
  float* wbuf = misc + 8192;           // up to N1*K (12984)
  float* zbuf = misc + 8192 + 16384;   // B*64 latent
  _Float16* wreg = (_Float16*)(misc + 32768);

  hipStream_t s = stream;

  // ---- Pack all weights to zero-padded f16 ----
  static const int ci[16] = {4,16,16,64,64,128,128,256,256,256,128,128,64,64,16,16};
  static const int co[16] = {16,16,64,64,128,128,256,256,256,128,128,64,64,16,16,4};
  const _Float16* Wp[32];
  size_t woff = 0;
  for (int l = 0; l < 16; ++l) {
    int Kp = SP(ci[l]), Np = SP(co[l]);
    int tot = Np * Kp;
    for (int h = 0; h < 2; ++h) {            // _l then _r
      _Float16* dst = wreg + woff;
      k_pack_w<<<nblk(tot), 256, 0, s>>>(Wc[2 * l + h], dst, co[l], ci[l], Kp, tot);
      Wp[2 * l + h] = dst;
      woff += tot;
    }
  }
  _Float16* W5p = wreg + woff;  woff += 64 * 512;
  k_pack_w<<<nblk(64 * 512), 256, 0, s>>>(W5, W5p, 64, 512, 512, 64 * 512);
  _Float16* W6p = wreg + woff;  woff += 512 * 64;
  k_pack_w<<<nblk(512 * 64), 256, 0, s>>>(W6, W6p, 512, 64, 64, 512 * 64);

  // ---------------- Encoder ----------------
  int totPad = BB * N1c * 32;
  k_normalize_pad<<<nblk(totPad), 256, 0, s>>>(x, bufA, totPad);

  run_sage(bufA, bufB, bufC, e1, E1c, N1c, 4, 16, Wp[0], Wp[1], deg, true, s);   // c1_1
  run_sage(bufC, bufB, bufA, e1, E1c, N1c, 16, 16, Wp[2], Wp[3], deg, true, s);  // c1_2
  run_pool(bufA, bufC, lab1, N1c, N2c, 16, deg, s);

  run_sage(bufC, bufB, bufA, e2, E2c, N2c, 16, 64, Wp[4], Wp[5], deg, true, s);  // c2_1
  run_sage(bufA, bufB, bufC, e2, E2c, N2c, 64, 64, Wp[6], Wp[7], deg, true, s);  // c2_2
  run_pool(bufC, bufA, lab2, N2c, N3c, 64, deg, s);

  run_sage(bufA, bufB, bufC, e3, E3c, N3c, 64, 128, Wp[8], Wp[9], deg, true, s);    // c3_1
  run_sage(bufC, bufB, bufA, e3, E3c, N3c, 128, 128, Wp[10], Wp[11], deg, true, s); // c3_2
  run_pool(bufA, bufC, lab3, N3c, N4c, 128, deg, s);

  run_sage(bufC, bufB, bufA, e4, E4c, N4c, 128, 256, Wp[12], Wp[13], deg, true, s); // c4_1
  run_sage(bufA, bufB, bufC, e4, E4c, N4c, 256, 256, Wp[14], Wp[15], deg, true, s); // c4_2
  run_pool(bufC, bufA, lab4, N4c, N5c, 256, deg, s);   // bufA = [B,2,256] = [B,512]

  // ---------------- Bottleneck dense ----------------
  run_gemm(bufA, W5p, nullptr, nullptr, b5, zbuf, BB, 512, 64, 64, false, s);  // z [B,64]
  run_gemm(zbuf, W6p, nullptr, nullptr, b6, bufB, BB, 64, 512, 512, true, s);  // h6 [B,512]

  // ---------------- Decoder ----------------
  run_knn(bufB, bufC, ai54, pos5, pos4, N5c, N4c, 256, deg, wbuf, s);
  run_sage(bufC, bufB, bufA, e4, E4c, N4c, 256, 256, Wp[16], Wp[17], deg, true, s); // c7_1
  run_sage(bufA, bufB, bufC, e4, E4c, N4c, 256, 256, Wp[18], Wp[19], deg, true, s); // c7_2

  run_knn(bufC, bufA, ai43, pos4, pos3, N4c, N3c, 256, deg, wbuf, s);
  run_sage(bufA, bufB, bufC, e3, E3c, N3c, 256, 128, Wp[20], Wp[21], deg, true, s); // c8_1
  run_sage(bufC, bufB, bufA, e3, E3c, N3c, 128, 128, Wp[22], Wp[23], deg, true, s); // c8_2

  run_knn(bufA, bufC, ai32, pos3, pos2, N3c, N2c, 128, deg, wbuf, s);
  run_sage(bufC, bufB, bufA, e2, E2c, N2c, 128, 64, Wp[24], Wp[25], deg, true, s);  // c9_1
  run_sage(bufA, bufB, bufC, e2, E2c, N2c, 64, 64, Wp[26], Wp[27], deg, true, s);   // c9_2

  run_knn(bufC, bufA, ai21, pos2, pos1, N2c, N1c, 64, deg, wbuf, s);
  run_sage(bufA, bufB, bufC, e1, E1c, N1c, 64, 16, Wp[28], Wp[29], deg, true, s);   // c10_1
  run_sage(bufC, bufB, bufA, e1, E1c, N1c, 16, 4, Wp[30], Wp[31], deg, false, s);   // c10_2

  int totOut = BB * N1c * 4;
  k_denorm<<<nblk(totOut), 256, 0, s>>>(bufA, (float*)d_out, totOut);
}
